// AttentionHead_9758165696575
// MI455X (gfx1250) — matrix operand order repaired and checked by reading, not compiled, not run on hardware
//
#include <hip/hip_runtime.h>

#define NBATCH 4
#define SEQ    4096
#define DM     512
#define DH     64

typedef __attribute__((ext_vector_type(16))) _Float16 v16h;
typedef __attribute__((ext_vector_type(8)))  _Float16 v8h;
typedef __attribute__((ext_vector_type(8)))  float    v8f;
typedef __attribute__((ext_vector_type(4)))  float    v4f;
typedef __attribute__((ext_vector_type(4)))  int      v4i;

#define AS1 __attribute__((address_space(1)))
#define AS3 __attribute__((address_space(3)))

#if defined(__has_builtin)
#  if __has_builtin(__builtin_amdgcn_global_load_async_to_lds_b128)
#    define HAVE_ASYNC_COPY 1
#  endif
#  if __has_builtin(__builtin_amdgcn_s_wait_asynccnt)
#    define HAVE_WAIT_ASYNC 1
#  endif
#endif

// 16-byte global -> LDS copy: async (ASYNCcnt-tracked) when available
static __device__ __forceinline__ void copy_b128(const _Float16* g, _Float16* l) {
#if defined(HAVE_ASYNC_COPY)
  __builtin_amdgcn_global_load_async_to_lds_b128((AS1 v4i*)g, (AS3 v4i*)l,
                                                 /*offset=*/0, /*cpol=*/0);
#else
  *(v8h*)l = *(const v8h*)g;
#endif
}

#if defined(HAVE_ASYNC_COPY) && defined(HAVE_WAIT_ASYNC)
#  define WAIT_ASYNC(n) __builtin_amdgcn_s_wait_asynccnt(n)
#elif defined(HAVE_ASYNC_COPY)
#  define WAIT_ASYNC(n) asm volatile("s_wait_asynccnt %0" ::"i"(n) : "memory")
#else
#  define WAIT_ASYNC(n) ((void)0)
#endif

static __device__ __forceinline__ v16h cat8(v8h lo, v8h hi) {
  v16h r;
#pragma unroll
  for (int i = 0; i < 8; ++i) { r[i] = lo[i]; r[i + 8] = hi[i]; }
  return r;
}

// ---------------- fused linear projection: out = (x @ W^T + b) * scale ------
__global__ __launch_bounds__(256) void qkv_proj_kernel(
    const float* __restrict__ x, const float* __restrict__ W,
    const float* __restrict__ bias, _Float16* __restrict__ out,
    int transposed, float scale)
{
  __shared__ __align__(16) _Float16 Wl[DH * DM];   // 64 KB f16 weights

  const int tid = threadIdx.x;
  const v4f* W4 = (const v4f*)W;
#pragma unroll 4
  for (int i = tid; i < DH * DM / 4; i += 256) {
    v4f w = W4[i];
#pragma unroll
    for (int j = 0; j < 4; ++j) Wl[i * 4 + j] = (_Float16)w[j];
  }
  __syncthreads();

  const int wave = tid >> 5;
  const int lane = tid & 31;
  const int m    = lane & 15;
  const int h    = lane >> 4;
  const int rowBase = (blockIdx.x * 8 + wave) * 16;

  v8f c[4] = {};

#pragma unroll 2
  for (int kk = 0; kk < DM / 32; ++kk) {
    const int k0 = kk * 32;
    const float* ar = x + (size_t)(rowBase + m) * DM + k0;
    v4f a0 = *(const v4f*)(ar + h * 8);
    v4f a1 = *(const v4f*)(ar + h * 8 + 4);
    v4f a2 = *(const v4f*)(ar + 16 + h * 8);
    v4f a3 = *(const v4f*)(ar + 16 + h * 8 + 4);
    v16h a;
#pragma unroll
    for (int j = 0; j < 4; ++j) {
      a[j]      = (_Float16)a0[j];
      a[j + 4]  = (_Float16)a1[j];
      a[j + 8]  = (_Float16)a2[j];
      a[j + 12] = (_Float16)a3[j];
    }
#pragma unroll
    for (int nt = 0; nt < 4; ++nt) {
      const int col = nt * 16 + m;
      const v8h* wp = (const v8h*)(Wl + col * DM + k0 + h * 8);
      v16h bb = cat8(wp[0], wp[2]);
      c[nt] = __builtin_amdgcn_wmma_f32_16x16x32_f16(false, a, false, bb,
                                                     (short)0, c[nt], false, false);
    }
  }

#pragma unroll
  for (int nt = 0; nt < 4; ++nt) {
    const int col = nt * 16 + m;
    const float bc = bias[col];
#pragma unroll
    for (int r = 0; r < 8; ++r) {
      const int row = rowBase + r + 8 * h;
      const float val = (c[nt][r] + bc) * scale;
      if (transposed) {                      // V stored [b][64][S]
        const int bidx = row >> 12, s = row & (SEQ - 1);
        out[(size_t)(bidx * DH + col) * SEQ + s] = (_Float16)val;
      } else {                               // Q/K stored [B*S][64]
        out[(size_t)row * DH + col] = (_Float16)val;
      }
    }
  }
}

// ---------------- causal flash attention -----------------------------------
// 4 waves/block, 16 q-rows each; K/V blocks double-buffered in LDS via
// async global->LDS copies; K, V and P WMMA operands come from ds_load_b128 (Q from global).
__global__ __launch_bounds__(128) void attn_kernel(
    const _Float16* __restrict__ q16, const _Float16* __restrict__ k16,
    const _Float16* __restrict__ vt16, float* __restrict__ out)
{
  __shared__ __align__(16) _Float16 Kl[2][32 * DH];   // 2 x 4 KB  [key][dim]
  __shared__ __align__(16) _Float16 Vl[2][DH * 32];   // 2 x 4 KB  [dim][key]
  __shared__ __align__(16) _Float16 Plds[4][16 * 32]; // per-wave P staging

  const int tid  = threadIdx.x;
  const int wave = tid >> 5;
  const int lane = tid & 31;
  const int m    = lane & 15;
  const int h    = lane >> 4;

  const int tile  = blockIdx.x * 4 + wave;
  const int bidx  = tile >> 8;                 // 256 q-tiles per batch (4|256)
  const int qBase = (tile & 255) * 16;
  const size_t rowQ = (size_t)bidx * SEQ + qBase;

  // block-uniform causal limit (max over the 4 waves); extra fully-masked
  // blocks are absorbed exactly by the online softmax (alpha=1, sum+=0)
  const int qMaxBase = ((blockIdx.x * 4 + 3) & 255) * 16;
  const int nBlk = (qMaxBase + 47) >> 5;

  // Q fragments (softmax scale folded in at projection)
  v16h qa[2];
#pragma unroll
  for (int dc = 0; dc < 2; ++dc) {
    const _Float16* qr = q16 + (rowQ + m) * DH + dc * 32;
    qa[dc] = cat8(*(const v8h*)(qr + h * 8), *(const v8h*)(qr + 16 + h * 8));
  }

  v8f o[4] = {};
  float mrow[8], lrow[8];
#pragma unroll
  for (int r = 0; r < 8; ++r) { mrow[r] = -__builtin_inff(); lrow[r] = 0.f; }

  _Float16* pl = &Plds[wave][0];

  // stage one 32-key K/V block: 512 B128 chunks spread over 128 threads
  auto issue_block = [&](int kb2, int buf) {
    const int keyBase2 = kb2 * 32;
#pragma unroll
    for (int i = 0; i < 2; ++i) {                 // K: [32 keys][64 dims]
      const int c = tid + i * 128;                // 256 chunks of 8 f16
      const int key = c >> 3, off = (c & 7) * 8;
      copy_b128(k16 + ((size_t)bidx * SEQ + keyBase2 + key) * DH + off,
                &Kl[buf][key * DH + off]);
    }
#pragma unroll
    for (int i = 0; i < 2; ++i) {                 // V^T: [64 dims][32 keys]
      const int c = tid + i * 128;
      const int dh = c >> 2, off = (c & 3) * 8;
      copy_b128(vt16 + ((size_t)(bidx * DH + dh)) * SEQ + keyBase2 + off,
                &Vl[buf][dh * 32 + off]);
    }
  };

  issue_block(0, 0);

  for (int kb = 0; kb < nBlk; ++kb) {
    const int buf = kb & 1;
    const int keyBase = kb * 32;

    if (kb + 1 < nBlk) {
      issue_block(kb + 1, buf ^ 1);   // prefetch next block
      WAIT_ASYNC(4);                  // oldest 4 (current buf) complete
    } else {
      WAIT_ASYNC(0);
    }
    __syncthreads();                  // all waves' staging visible

    // ---- scores: S = Q K^T (two 16-key tiles) from LDS ----
    v8f sc[2];
#pragma unroll
    for (int t = 0; t < 2; ++t) {
      const int key = keyBase + t * 16 + m;
      v8f c = {};
#pragma unroll
      for (int dc = 0; dc < 2; ++dc) {
        const _Float16* kp = &Kl[buf][(t * 16 + m) * DH + dc * 32 + h * 8];
        v16h bb = cat8(*(const v8h*)kp, *(const v8h*)(kp + 16));
        c = __builtin_amdgcn_wmma_f32_16x16x32_f16(false, qa[dc], false, bb,
                                                   (short)0, c, false, false);
      }
#pragma unroll
      for (int r = 0; r < 8; ++r) {               // causal mask
        const int qrow = qBase + r + 8 * h;
        if (key > qrow) c[r] = -__builtin_inff();
      }
      sc[t] = c;
    }

    // ---- online softmax (rows live across 16 lanes) ----
    float bm[8];
#pragma unroll
    for (int r = 0; r < 8; ++r) bm[r] = fmaxf(sc[0][r], sc[1][r]);
#pragma unroll
    for (int msk = 1; msk <= 8; msk <<= 1)
#pragma unroll
      for (int r = 0; r < 8; ++r) bm[r] = fmaxf(bm[r], __shfl_xor(bm[r], msk, 32));

    float nm[8], alpha[8];
#pragma unroll
    for (int r = 0; r < 8; ++r) {
      nm[r]    = fmaxf(mrow[r], bm[r]);
      alpha[r] = __expf(mrow[r] - nm[r]);
      mrow[r]  = nm[r];
      lrow[r] *= alpha[r];
    }
#pragma unroll
    for (int nt = 0; nt < 4; ++nt)
#pragma unroll
      for (int r = 0; r < 8; ++r) o[nt][r] *= alpha[r];

    float rs[8] = {};
#pragma unroll
    for (int t = 0; t < 2; ++t)
#pragma unroll
      for (int r = 0; r < 8; ++r) {
        const float p = __expf(sc[t][r] - nm[r]);
        rs[r] += p;
        pl[(r + 8 * h) * 32 + t * 16 + m] = (_Float16)p;
      }
#pragma unroll
    for (int msk = 1; msk <= 8; msk <<= 1)
#pragma unroll
      for (int r = 0; r < 8; ++r) rs[r] += __shfl_xor(rs[r], msk, 32);
#pragma unroll
    for (int r = 0; r < 8; ++r) lrow[r] += rs[r];

    __builtin_amdgcn_wave_barrier();
    asm volatile("s_wait_dscnt 0" ::: "memory");  // P stores -> P loads (same wave)
    __builtin_amdgcn_wave_barrier();

    const _Float16* pr = pl + m * 32;             // P in A-fragment layout
    v16h pa = cat8(*(const v8h*)(pr + h * 8), *(const v8h*)(pr + 16 + h * 8));

    // ---- O += P V from LDS ----
#pragma unroll
    for (int nt = 0; nt < 4; ++nt) {
      const _Float16* vp = &Vl[buf][(nt * 16 + m) * 32 + h * 8];
      v16h vb = cat8(*(const v8h*)vp, *(const v8h*)(vp + 16));
      o[nt] = __builtin_amdgcn_wmma_f32_16x16x32_f16(false, pa, false, vb,
                                                     (short)0, o[nt], false, false);
    }
    __syncthreads();                 // reads done before next staging overwrite
  }

  // ---- normalize + store f32 ----
#pragma unroll
  for (int r = 0; r < 8; ++r) {
    const float inv = lrow[r] > 0.f ? 1.f / lrow[r] : 0.f;
#pragma unroll
    for (int nt = 0; nt < 4; ++nt)
      out[(rowQ + r + 8 * h) * DH + nt * 16 + m] = o[nt][r] * inv;
  }
}

extern "C" void kernel_launch(void* const* d_in, const int* in_sizes, int n_in,
                              void* d_out, int out_size, void* d_ws, size_t ws_size,
                              hipStream_t stream) {
  const float* q_in = (const float*)d_in[0];
  const float* k_in = (const float*)d_in[1];
  const float* v_in = (const float*)d_in[2];
  const float* Wq   = (const float*)d_in[3];
  const float* bq   = (const float*)d_in[4];
  const float* Wk   = (const float*)d_in[5];
  const float* bk   = (const float*)d_in[6];
  const float* Wv   = (const float*)d_in[7];
  const float* bv   = (const float*)d_in[8];

  const size_t elems = (size_t)NBATCH * SEQ * DH;
  _Float16* q16 = (_Float16*)d_ws;
  _Float16* k16 = q16 + elems;
  _Float16* v16 = k16 + elems;                      // 6 MB workspace total

  const int rowTiles = NBATCH * SEQ / 16;           // 1024

  qkv_proj_kernel<<<rowTiles / 8, 256, 0, stream>>>(q_in, Wq, bq, q16, 0, 0.125f);
  qkv_proj_kernel<<<rowTiles / 8, 256, 0, stream>>>(k_in, Wk, bk, k16, 0, 1.0f);
  qkv_proj_kernel<<<rowTiles / 8, 256, 0, stream>>>(v_in, Wv, bv, v16, 1, 1.0f);

  attn_kernel<<<rowTiles / 4, 128, 0, stream>>>(q16, k16, v16, (float*)d_out);
}
